// GCN5_1_83227876262529
// MI455X (gfx1250) — compile-verified, matrix operand order checked by reading, not run on hardware
//
#include <hip/hip_runtime.h>
#include <hip/hip_bf16.h>
#include <math.h>

// ---------------------------------------------------------------------------
// Types for CDNA5 WMMA (wave32, V_WMMA_F32_16X16X32_BF16)
// ---------------------------------------------------------------------------
typedef __attribute__((ext_vector_type(16))) __bf16         v16bf;
typedef __attribute__((ext_vector_type(16))) unsigned short v16u;
typedef __attribute__((ext_vector_type(8)))  unsigned short v8us;
typedef __attribute__((ext_vector_type(8)))  float          v8f;

__device__ __forceinline__ unsigned short f32_to_bf16(float f) {
    unsigned u = __float_as_uint(f);
    unsigned r = u + 0x7FFFu + ((u >> 16) & 1u);   // round-to-nearest-even
    return (unsigned short)(r >> 16);
}

// order-preserving float <-> uint mapping for atomicMax-based segment max
__device__ __forceinline__ unsigned f2ord(float f) {
    unsigned u = __float_as_uint(f);
    return (u & 0x80000000u) ? ~u : (u | 0x80000000u);
}
__device__ __forceinline__ float ord2f(unsigned u) {
    unsigned v = (u & 0x80000000u) ? (u & 0x7FFFFFFFu) : ~u;
    return __uint_as_float(v);
}

__device__ __forceinline__ void edge_sd(const int* __restrict__ ei, int E0,
                                        long long e, int& s, int& d) {
    if (e < (long long)E0) { s = ei[e]; d = ei[(long long)E0 + e]; }
    else                   { s = d = (int)(e - E0); }   // self-loop
}

// ---------------------------------------------------------------------------
// Operand prep: f32 -> bf16, zero-padded to Kpad; W additionally transposed.
// ---------------------------------------------------------------------------
__global__ void k_cvt_a_bf16(const float* __restrict__ A,
                             unsigned short* __restrict__ Abf,
                             long long total, int K, int Kpad) {
    for (long long t = blockIdx.x * (long long)blockDim.x + threadIdx.x; t < total;
         t += (long long)gridDim.x * blockDim.x) {
        long long n = t / Kpad; int k = (int)(t % Kpad);
        Abf[t] = (k < K) ? f32_to_bf16(A[(size_t)n * K + k]) : (unsigned short)0;
    }
}

__global__ void k_cvt_wt_bf16(const float* __restrict__ W,
                              unsigned short* __restrict__ Wt,
                              long long total, int K, int Kpad, int Nc) {
    // Wt[n * Kpad + k] = W[k * Nc + n], zero padded in both dims
    for (long long t = blockIdx.x * (long long)blockDim.x + threadIdx.x; t < total;
         t += (long long)gridDim.x * blockDim.x) {
        long long n = t / Kpad; int k = (int)(t % Kpad);
        Wt[t] = (k < K && n < Nc) ? f32_to_bf16(W[(size_t)k * Nc + n])
                                  : (unsigned short)0;
    }
}

// ---------------------------------------------------------------------------
// GEMM: C[M,Nc] = A[M,K] * W[K,Nc] using padded bf16 operands.
// One wave = one 16x16 tile; 4 waves/block. Branch-free b128 fragment loads.
// Fragment layouts per CDNA5 ISA 7.12.2 (16-bit A 16x32, f32 C/D 16x16):
//   lanes 0-15  : row/col = lane,    halves 0-7 = K kb..kb+7, 8-15 = kb+16..kb+23
//   lanes 16-31 : row/col = lane-16, kb offset by +8 (K 8-15 / 24-31)
// ---------------------------------------------------------------------------
__device__ __forceinline__ v16bf load_frag(const unsigned short* __restrict__ p,
                                           int kb) {
    v8us lo = *reinterpret_cast<const v8us*>(p + kb);
    v8us hi = *reinterpret_cast<const v8us*>(p + kb + 16);
    v16u u = __builtin_shufflevector(lo, hi, 0, 1, 2, 3, 4, 5, 6, 7,
                                              8, 9, 10, 11, 12, 13, 14, 15);
    return __builtin_bit_cast(v16bf, u);
}

__global__ void k_gemm_wmma_bf16(const unsigned short* __restrict__ Abf,
                                 const unsigned short* __restrict__ Wt,
                                 float* __restrict__ C,
                                 int M, int Kpad, int Nc) {
    const int wave  = threadIdx.x >> 5;
    const int lane  = threadIdx.x & 31;
    const int tileM = (blockIdx.x * (blockDim.x >> 5) + wave) * 16;
    const int tileN = blockIdx.y * 16;
    if (tileM >= M) return;                    // wave-uniform: EXEC stays all-1

    const int l15  = lane & 15;
    const int hi   = lane >> 4;                // K sub-pattern select
    const int rowA = tileM + l15;              // always < M (M % 16 == 0)
    const int colB = tileN + l15;              // always < Ncpad (padded Wt)

    const unsigned short* ap = Abf + (size_t)rowA * Kpad;
    const unsigned short* bp = Wt  + (size_t)colB * Kpad;
    const int kofs = hi * 8;

    v8f acc0 = {}, acc1 = {};
    int k0 = 0;
    for (; k0 + 64 <= Kpad; k0 += 64) {
        v16bf a0 = load_frag(ap, k0 + kofs);
        v16bf b0 = load_frag(bp, k0 + kofs);
        v16bf a1 = load_frag(ap, k0 + 32 + kofs);
        v16bf b1 = load_frag(bp, k0 + 32 + kofs);
        acc0 = __builtin_amdgcn_wmma_f32_16x16x32_bf16(
                   false, a0, false, b0, (short)0, acc0, false, false);
        acc1 = __builtin_amdgcn_wmma_f32_16x16x32_bf16(
                   false, a1, false, b1, (short)0, acc1, false, false);
    }
    if (k0 < Kpad) {
        v16bf a0 = load_frag(ap, k0 + kofs);
        v16bf b0 = load_frag(bp, k0 + kofs);
        acc0 = __builtin_amdgcn_wmma_f32_16x16x32_bf16(
                   false, a0, false, b0, (short)0, acc0, false, false);
    }
    v8f acc = acc0 + acc1;

    // C/D layout: lane = N column; VGPR r holds M=r (lanes 0-15), M=r+8 (16-31)
    const int  rb    = hi * 8;
    const bool colOK = colB < Nc;
#pragma unroll
    for (int r = 0; r < 8; ++r) {
        int row = tileM + rb + r;
        if (row < M && colOK) C[(size_t)row * Nc + colB] = acc[r];
    }
}

// ---------------------------------------------------------------------------
// Attention coefficients: als[n,h] = sum_c h[n,h,c]*a_s[h,c] (and ald)
// ---------------------------------------------------------------------------
__global__ void k_attn(const float* __restrict__ H, const float* __restrict__ as_,
                       const float* __restrict__ ad_, float* __restrict__ als,
                       float* __restrict__ ald, long long NH, int Hh, int Cc) {
    for (long long t = blockIdx.x * (long long)blockDim.x + threadIdx.x; t < NH;
         t += (long long)gridDim.x * blockDim.x) {
        long long n = t / Hh; int h = (int)(t % Hh);
        const float* hp = H + ((size_t)n * Hh + h) * Cc;
        float s1 = 0.f, s2 = 0.f;
        for (int c = 0; c < Cc; ++c) {
            float v = hp[c];
            s1 += v * as_[h * Cc + c];
            s2 += v * ad_[h * Cc + c];
        }
        als[t] = s1; ald[t] = s2;
    }
}

__global__ void k_init_seg(unsigned* __restrict__ maxb, float* __restrict__ sumb,
                           long long total) {
    const unsigned NEG_INF_ORD = f2ord(-INFINITY);
    for (long long t = blockIdx.x * (long long)blockDim.x + threadIdx.x; t < total;
         t += (long long)gridDim.x * blockDim.x) {
        maxb[t] = NEG_INF_ORD; sumb[t] = 0.f;
    }
}

__global__ void k_edge_logits(const int* __restrict__ ei, int E0, long long EH,
                              const float* __restrict__ als,
                              const float* __restrict__ ald,
                              float* __restrict__ ebuf,
                              unsigned* __restrict__ maxb, int Hh) {
    for (long long t = blockIdx.x * (long long)blockDim.x + threadIdx.x; t < EH;
         t += (long long)gridDim.x * blockDim.x) {
        long long e = t / Hh; int h = (int)(t % Hh);
        int s, d; edge_sd(ei, E0, e, s, d);
        float v = als[(size_t)s * Hh + h] + ald[(size_t)d * Hh + h];
        v = (v > 0.f) ? v : 0.2f * v;                 // leaky_relu(NEG_SLOPE)
        ebuf[t] = v;
        atomicMax(&maxb[(size_t)d * Hh + h], f2ord(v));
    }
}

__global__ void k_edge_exp(const int* __restrict__ ei, int E0, long long EH,
                           const unsigned* __restrict__ maxb,
                           float* __restrict__ ebuf, float* __restrict__ sumb,
                           int Hh) {
    for (long long t = blockIdx.x * (long long)blockDim.x + threadIdx.x; t < EH;
         t += (long long)gridDim.x * blockDim.x) {
        long long e = t / Hh; int h = (int)(t % Hh);
        int s, d; edge_sd(ei, E0, e, s, d);
        float m = ord2f(maxb[(size_t)d * Hh + h]);
        if (!__builtin_isfinite(m)) m = 0.f;          // ref: where(isfinite)
        float ev = expf(ebuf[t] - m);
        ebuf[t] = ev;
        atomicAdd(&sumb[(size_t)d * Hh + h], ev);
    }
}

__global__ void k_edge_alpha(const int* __restrict__ ei, int E0, long long EH,
                             const float* __restrict__ sumb,
                             float* __restrict__ ebuf, int Hh) {
    for (long long t = blockIdx.x * (long long)blockDim.x + threadIdx.x; t < EH;
         t += (long long)gridDim.x * blockDim.x) {
        long long e = t / Hh; int h = (int)(t % Hh);
        int s, d; edge_sd(ei, E0, e, s, d);
        ebuf[t] = ebuf[t] / (sumb[(size_t)d * Hh + h] + 1e-16f);
    }
}

__global__ void k_init_bias(float* __restrict__ O, const float* __restrict__ b,
                            long long total, int F) {
    for (long long t = blockIdx.x * (long long)blockDim.x + threadIdx.x; t < total;
         t += (long long)gridDim.x * blockDim.x)
        O[t] = b ? b[t % F] : 0.f;
}

// aggregation: 4 features per thread (F % 4 == 0 in every layer)
__global__ void k_edge_agg4(const int* __restrict__ ei, int E0, long long EQ,
                            const float* __restrict__ H,
                            const float* __restrict__ alpha,
                            float* __restrict__ O, int F, int Cc, int Hh) {
    const int Q = F >> 2;
    for (long long t = blockIdx.x * (long long)blockDim.x + threadIdx.x; t < EQ;
         t += (long long)gridDim.x * blockDim.x) {
        long long e = t / Q; int f0 = (int)(t % Q) << 2;
        int s, d; edge_sd(ei, E0, e, s, d);
        const float4 hv = *reinterpret_cast<const float4*>(H + (size_t)s * F + f0);
        const float* al = alpha + (size_t)e * Hh;
        float* op = O + (size_t)d * F + f0;
        atomicAdd(op + 0, hv.x * al[(f0 + 0) / Cc]);
        atomicAdd(op + 1, hv.y * al[(f0 + 1) / Cc]);
        atomicAdd(op + 2, hv.z * al[(f0 + 2) / Cc]);
        atomicAdd(op + 3, hv.w * al[(f0 + 3) / Cc]);
    }
}

__global__ void k_relu(float* __restrict__ X, long long total) {
    for (long long t = blockIdx.x * (long long)blockDim.x + threadIdx.x; t < total;
         t += (long long)gridDim.x * blockDim.x)
        X[t] = fmaxf(X[t], 0.f);
}

__global__ void k_bn_zero(float* __restrict__ s, float* __restrict__ q, int F) {
    int t = blockIdx.x * blockDim.x + threadIdx.x;
    if (t < F) { s[t] = 0.f; q[t] = 0.f; }
}

__global__ void k_bn_acc(const float* __restrict__ X, float* __restrict__ s,
                         float* __restrict__ q, long long total, int F) {
    for (long long t = blockIdx.x * (long long)blockDim.x + threadIdx.x; t < total;
         t += (long long)gridDim.x * blockDim.x) {
        float v = X[t]; int f = (int)(t % F);
        atomicAdd(&s[f], v);
        atomicAdd(&q[f], v * v);
    }
}

__global__ void k_bn_apply(float* __restrict__ X, const float* __restrict__ s,
                           const float* __restrict__ q, const float* __restrict__ g,
                           const float* __restrict__ be, long long total, int F,
                           float invN) {
    for (long long t = blockIdx.x * (long long)blockDim.x + threadIdx.x; t < total;
         t += (long long)gridDim.x * blockDim.x) {
        int f = (int)(t % F);
        float mu  = s[f] * invN;
        float var = q[f] * invN - mu * mu;
        float y = (X[t] - mu) * rsqrtf(var + 1e-5f) * g[f] + be[f];
        X[t] = (y > 0.f) ? y : 0.1f * y;              // leaky_relu(0.1)
    }
}

__global__ void k_final(const float* __restrict__ A, const float* __restrict__ b32,
                        float* __restrict__ out, int Nn, int Hh) {
    for (long long t = blockIdx.x * (long long)blockDim.x + threadIdx.x; t < Nn;
         t += (long long)gridDim.x * blockDim.x) {
        float sum = 0.f;
        for (int h = 0; h < Hh; ++h) sum += A[(size_t)t * Hh + h];
        float v = sum / (float)Hh + b32[0];
        out[t] = 1.f / (1.f + expf(-v));
    }
}

// ---------------------------------------------------------------------------
// Host-side orchestration
// ---------------------------------------------------------------------------
static inline unsigned nblk(long long total, int tpb) {
    long long b = (total + tpb - 1) / tpb;
    if (b > 1048576) b = 1048576;
    if (b < 1) b = 1;
    return (unsigned)b;
}

static void run_conv(hipStream_t stream, const float* X, int Fin, const float* W,
                     const float* a_s, const float* a_d, const float* bias,
                     int Hh, int Cc, const int* ei, int E0, int Nn,
                     unsigned short* abf, unsigned short* wtbf,
                     float* Hbuf, float* Obuf, float* als, float* ald,
                     unsigned* maxb, float* sumb, float* ebuf) {
    const int F     = Hh * Cc;
    const int Kpad  = (Fin + 31) & ~31;
    const int Ncpad = (F + 15) & ~15;
    const long long ET = (long long)E0 + Nn;
    const long long NH = (long long)Nn * Hh;
    const long long EH = ET * Hh;
    const long long EQ = ET * (F >> 2);
    const long long NF = (long long)Nn * F;

    // operand prep (bf16, padded; W transposed)
    long long aTot = (long long)Nn * Kpad;
    long long wTot = (long long)Ncpad * Kpad;
    k_cvt_a_bf16<<<nblk(aTot, 256), 256, 0, stream>>>(X, abf, aTot, Fin, Kpad);
    k_cvt_wt_bf16<<<nblk(wTot, 256), 256, 0, stream>>>(W, wtbf, wTot, Fin, Kpad, F);

    // h = X @ W  (bf16 WMMA)
    int mtiles = (Nn + 15) / 16;
    dim3 gg((mtiles + 3) / 4, Ncpad / 16);
    k_gemm_wmma_bf16<<<gg, 128, 0, stream>>>(abf, wtbf, Hbuf, Nn, Kpad, F);

    k_attn<<<nblk(NH, 256), 256, 0, stream>>>(Hbuf, a_s, a_d, als, ald, NH, Hh, Cc);
    k_init_seg<<<nblk(NH, 256), 256, 0, stream>>>(maxb, sumb, NH);
    k_edge_logits<<<nblk(EH, 256), 256, 0, stream>>>(ei, E0, EH, als, ald, ebuf, maxb, Hh);
    k_edge_exp<<<nblk(EH, 256), 256, 0, stream>>>(ei, E0, EH, maxb, ebuf, sumb, Hh);
    k_edge_alpha<<<nblk(EH, 256), 256, 0, stream>>>(ei, E0, EH, sumb, ebuf, Hh);
    k_init_bias<<<nblk(NF, 256), 256, 0, stream>>>(Obuf, bias, NF, F);
    k_edge_agg4<<<nblk(EQ, 256), 256, 0, stream>>>(ei, E0, EQ, Hbuf, ebuf, Obuf, F, Cc, Hh);
}

extern "C" void kernel_launch(void* const* d_in, const int* in_sizes, int n_in,
                              void* d_out, int out_size, void* d_ws, size_t ws_size,
                              hipStream_t stream) {
    const float* x  = (const float*)d_in[0];
    const int*   ei = (const int*)d_in[1];
    const int Nn = in_sizes[0] / 3;
    const int E0 = in_sizes[1] / 2;

    const float *W11=(const float*)d_in[2],  *as11=(const float*)d_in[3],
                *ad11=(const float*)d_in[4], *b11=(const float*)d_in[5];
    const float *W12=(const float*)d_in[6],  *as12=(const float*)d_in[7],
                *ad12=(const float*)d_in[8], *b12=(const float*)d_in[9];
    const float *g1=(const float*)d_in[10],  *be1=(const float*)d_in[11];
    const float *W21=(const float*)d_in[12], *as21=(const float*)d_in[13],
                *ad21=(const float*)d_in[14],*b21=(const float*)d_in[15];
    const float *W22=(const float*)d_in[16], *as22=(const float*)d_in[17],
                *ad22=(const float*)d_in[18],*b22=(const float*)d_in[19];
    const float *g2=(const float*)d_in[20],  *be2=(const float*)d_in[21];
    const float *W31=(const float*)d_in[22], *as31=(const float*)d_in[23],
                *ad31=(const float*)d_in[24],*b31=(const float*)d_in[25];
    const float *W32=(const float*)d_in[26], *as32=(const float*)d_in[27],
                *ad32=(const float*)d_in[28],*b32=(const float*)d_in[29];

    // workspace carve-out
    char* w = (char*)d_ws;
    size_t off = 0;
    auto alloc = [&](size_t bytes) {
        size_t o = off; off += (bytes + 255) & ~(size_t)255; return o;
    };
    const size_t nodeB = (size_t)Nn * 512 * sizeof(float);
    float*    buf0 = (float*)(w + alloc(nodeB));
    float*    buf1 = (float*)(w + alloc(nodeB));
    float*    bufH = (float*)(w + alloc(nodeB));
    unsigned short* abf  = (unsigned short*)(w + alloc((size_t)Nn * 512 * 2));
    unsigned short* wtbf = (unsigned short*)(w + alloc((size_t)512 * 512 * 2));
    float*    als  = (float*)(w + alloc((size_t)Nn * 8 * sizeof(float)));
    float*    ald  = (float*)(w + alloc((size_t)Nn * 8 * sizeof(float)));
    unsigned* maxb = (unsigned*)(w + alloc((size_t)Nn * 8 * sizeof(unsigned)));
    float*    sumb = (float*)(w + alloc((size_t)Nn * 8 * sizeof(float)));
    float*    ebuf = (float*)(w + alloc(((size_t)E0 + Nn) * 8 * sizeof(float)));
    float*    bns  = (float*)(w + alloc(512 * sizeof(float)));
    float*    bnq  = (float*)(w + alloc(512 * sizeof(float)));

    const float invN = 1.f / (float)Nn;

    // ---- Block 1: GAT(3 -> 64, heads=8, C=8) ----
    run_conv(stream, x, 3, W11, as11, ad11, b11, 8, 8, ei, E0, Nn,
             abf, wtbf, bufH, buf0, als, ald, maxb, sumb, ebuf);
    k_relu<<<nblk((long long)Nn * 64, 256), 256, 0, stream>>>(buf0, (long long)Nn * 64);

    run_conv(stream, buf0, 64, W12, as12, ad12, b12, 8, 8, ei, E0, Nn,
             abf, wtbf, bufH, buf1, als, ald, maxb, sumb, ebuf);
    k_bn_zero<<<1, 64, 0, stream>>>(bns, bnq, 64);
    k_bn_acc<<<nblk((long long)Nn * 64, 256), 256, 0, stream>>>(buf1, bns, bnq, (long long)Nn * 64, 64);
    k_bn_apply<<<nblk((long long)Nn * 64, 256), 256, 0, stream>>>(buf1, bns, bnq, g1, be1, (long long)Nn * 64, 64, invN);

    // ---- Block 2: GAT(64 -> 256, heads=8, C=32) ----
    run_conv(stream, buf1, 64, W21, as21, ad21, b21, 8, 32, ei, E0, Nn,
             abf, wtbf, bufH, buf0, als, ald, maxb, sumb, ebuf);
    k_relu<<<nblk((long long)Nn * 256, 256), 256, 0, stream>>>(buf0, (long long)Nn * 256);

    run_conv(stream, buf0, 256, W22, as22, ad22, b22, 8, 32, ei, E0, Nn,
             abf, wtbf, bufH, buf1, als, ald, maxb, sumb, ebuf);
    k_bn_zero<<<1, 256, 0, stream>>>(bns, bnq, 256);
    k_bn_acc<<<nblk((long long)Nn * 256, 256), 256, 0, stream>>>(buf1, bns, bnq, (long long)Nn * 256, 256);
    k_bn_apply<<<nblk((long long)Nn * 256, 256), 256, 0, stream>>>(buf1, bns, bnq, g2, be2, (long long)Nn * 256, 256, invN);

    // ---- Block 3: GAT(256 -> 512, heads=8, C=64) then (512 -> 1, mean heads) ----
    run_conv(stream, buf1, 256, W31, as31, ad31, b31, 8, 64, ei, E0, Nn,
             abf, wtbf, bufH, buf0, als, ald, maxb, sumb, ebuf);
    k_relu<<<nblk((long long)Nn * 512, 256), 256, 0, stream>>>(buf0, (long long)Nn * 512);

    run_conv(stream, buf0, 512, W32, as32, ad32, nullptr, 8, 1, ei, E0, Nn,
             abf, wtbf, bufH, buf1, als, ald, maxb, sumb, ebuf);
    k_final<<<nblk(Nn, 256), 256, 0, stream>>>(buf1, b32, (float*)d_out, Nn, 8);

    (void)n_in; (void)out_size; (void)ws_size;
}
